// Decoder_68693706932595
// MI455X (gfx1250) — matrix operand order repaired and checked by reading, not compiled, not run on hardware
//
#include <hip/hip_runtime.h>
#include <hip/hip_bf16.h>

typedef __attribute__((ext_vector_type(16))) __bf16 v16bf;
typedef __attribute__((ext_vector_type(8)))  __bf16 v8bf;
typedef __attribute__((ext_vector_type(8)))  float  v8f;

#define B_DIM   4096
#define H_DIM   32
#define F_DIM   1024
#define NCAT    8
#define CATD    16
#define NCONT   24

#define BM 128
#define BN 128
#define BK 32
#define KT (F_DIM / BK)     // 32 k-tiles
#define LDA 40              // padded LDS row (halfs) = 80B, 16B-aligned offsets

#define CAT_OFF  0
#define CONT_OFF (B_DIM * NCAT * CATD)          // 524288
#define LAT_OFF  (CONT_OFF + B_DIM * NCONT)     // 622592

#if __has_builtin(__builtin_amdgcn_cvt_pk_bf16_f32)
typedef __attribute__((ext_vector_type(2))) __bf16 v2bf;
__device__ __forceinline__ unsigned pack_bf16x2(float a, float b) {
  v2bf r = __builtin_amdgcn_cvt_pk_bf16_f32(a, b);
  return __builtin_bit_cast(unsigned, r);
}
#else
__device__ __forceinline__ unsigned pack_bf16x2(float a, float b) {
  unsigned ua = __float_as_uint(a);
  unsigned ub = __float_as_uint(b);
  ua = (ua + 0x7FFFu + ((ua >> 16) & 1u)) >> 16;   // RNE f32 -> bf16
  ub = (ub + 0x7FFFu + ((ub >> 16) & 1u)) >> 16;
  return (ua & 0xFFFFu) | (ub << 16);
}
#endif

struct Fetch {
  float2 a[8];   // A tile: 128 rows x 32 cols, 16 threads/row, 8 sweeps
  float2 b[8];   // B tile: 32 k-rows x 128 cols, 8 threads/row, 8 col sweeps
};

// latent_raw[b][h][d] = sum_f z[b][f] * Wp[h][f][d] + bp[h][d]
__global__ __launch_bounds__(256)
void gemm_proj_kernel(const float* __restrict__ z,
                      const float* __restrict__ Wp,
                      const float* __restrict__ bp,
                      float* __restrict__ lat) {
  __shared__ __align__(16) unsigned short As[2 * BM * LDA];  // 20.0 KB
  __shared__ __align__(16) unsigned short Bs[2 * BN * LDA];  // 20.0 KB (B stored [n][k])

  const int tid  = threadIdx.x;
  const int h    = blockIdx.z;
  const int m0   = blockIdx.y * BM;
  const int n0   = blockIdx.x * BN;
  const int lane = tid & 31;
  const int w    = tid >> 5;
  const int wm   = w & 3;      // wave tile row (4 over M, 32 rows each)
  const int wn   = w >> 2;     // wave tile col (2 over N, 64 cols each)

  const int arow = tid >> 4;          // 0..15, +16 per sweep
  const int acol = (tid & 15) * 2;    // half index within k-tile
  const int bk   = tid >> 3;          // 0..31 (full k range)
  const int bn2  = tid & 7;

  const float* zbase = z + (size_t)m0 * F_DIM;
  const float* wbase = Wp + (size_t)h * F_DIM * F_DIM;

  auto fetch = [&](int kt, Fetch& f) {
    const int k0 = kt * BK;
#pragma unroll
    for (int s = 0; s < 8; ++s) {
      const float* p = zbase + (size_t)(arow + s * 16) * F_DIM + (k0 + acol);
      f.a[s] = *reinterpret_cast<const float2*>(p);
    }
#pragma unroll
    for (int s = 0; s < 8; ++s) {
      const int n = (s * 8 + bn2) * 2;
      const float* p = wbase + (size_t)(k0 + bk) * F_DIM + (n0 + n);
      f.b[s] = *reinterpret_cast<const float2*>(p);
    }
  };

  auto stage = [&](int buf, const Fetch& f) {
    unsigned* a32 = reinterpret_cast<unsigned*>(As);
#pragma unroll
    for (int s = 0; s < 8; ++s) {
      const int row = arow + s * 16;
      a32[((buf * BM + row) * LDA + acol) >> 1] = pack_bf16x2(f.a[s].x, f.a[s].y);
    }
#pragma unroll
    for (int s = 0; s < 8; ++s) {        // transpose B into [n][k]
      const int n = (s * 8 + bn2) * 2;
      const unsigned pk = pack_bf16x2(f.b[s].x, f.b[s].y);
      Bs[(buf * BN + n    ) * LDA + bk] = (unsigned short)(pk & 0xFFFFu);
      Bs[(buf * BN + n + 1) * LDA + bk] = (unsigned short)(pk >> 16);
    }
  };

  // ISA 16-bit A layout: lanes 0-15 hold K {kh*8..+7, 16+kh*8..+7} for kh=lane>>4
  auto load_frag_a = [&](int buf, int mg) -> v16bf {
    const int m  = wm * 32 + mg * 16 + (lane & 15);
    const int kh = lane >> 4;
    const v8bf lo = *reinterpret_cast<const v8bf*>(&As[(buf * BM + m) * LDA + kh * 8]);
    const v8bf hi = *reinterpret_cast<const v8bf*>(&As[(buf * BM + m) * LDA + 16 + kh * 8]);
    return __builtin_shufflevector(lo, hi, 0,1,2,3,4,5,6,7,8,9,10,11,12,13,14,15);
  };
  // B column-major: lane holds one column's K {kh*8..+7, 16+kh*8..+7} (matches A)
  auto load_frag_b = [&](int buf, int ng) -> v16bf {
    const int n  = wn * 64 + ng * 16 + (lane & 15);
    const int kh = lane >> 4;
    const v8bf lo = *reinterpret_cast<const v8bf*>(&Bs[(buf * BN + n) * LDA + kh * 8]);
    const v8bf hi = *reinterpret_cast<const v8bf*>(&Bs[(buf * BN + n) * LDA + 16 + kh * 8]);
    return __builtin_shufflevector(lo, hi, 0,1,2,3,4,5,6,7,8,9,10,11,12,13,14,15);
  };

  v8f acc[2][4] = {};

  Fetch f;
  fetch(0, f);
  stage(0, f);
  __syncthreads();

  for (int kt = 0; kt < KT; ++kt) {
    const int buf = kt & 1;
    if (kt + 1 < KT) fetch(kt + 1, f);   // global loads overlap WMMA below

    const v16bf a0 = load_frag_a(buf, 0);
    const v16bf a1 = load_frag_a(buf, 1);
    v16bf bfr[4];
#pragma unroll
    for (int ng = 0; ng < 4; ++ng) bfr[ng] = load_frag_b(buf, ng);

#pragma unroll
    for (int ng = 0; ng < 4; ++ng) {
      acc[0][ng] = __builtin_amdgcn_wmma_f32_16x16x32_bf16(false, a0, false, bfr[ng], (short)0, acc[0][ng], false, false);
      acc[1][ng] = __builtin_amdgcn_wmma_f32_16x16x32_bf16(false, a1, false, bfr[ng], (short)0, acc[1][ng], false, false);
    }

    if (kt + 1 < KT) stage((kt + 1) & 1, f);
    __syncthreads();
  }

  // C/D layout: lanes 0-15: VGPR r = row r; lanes 16-31: VGPR r = row 8+r; col = lane&15
  const int rsel = (lane >> 4) * 8;
  const int ncol = lane & 15;
#pragma unroll
  for (int mg = 0; mg < 2; ++mg) {
#pragma unroll
    for (int ng = 0; ng < 4; ++ng) {
      const int col = n0 + wn * 64 + ng * 16 + ncol;
      const float bias = bp[h * F_DIM + col];
#pragma unroll
      for (int r = 0; r < 8; ++r) {
        const int row = m0 + wm * 32 + mg * 16 + rsel + r;
        lat[((size_t)row * H_DIM + h) * F_DIM + col] = acc[mg][ng][r] + bias;
      }
    }
  }
}

// One block per batch row b: LayerNorm over [H,F] + ReLU + softmax/scalar heads,
// with the full 128KB row cached in LDS (320KB/WGP -> 2 blocks resident).
__global__ __launch_bounds__(256)
void ln_head_kernel(float* __restrict__ out,
                    const float* __restrict__ ln_w,
                    const float* __restrict__ ln_b,
                    const float* __restrict__ Wc,
                    const float* __restrict__ bc,
                    const float* __restrict__ Wv,
                    const float* __restrict__ bv) {
  extern __shared__ float srow[];     // H_DIM*F_DIM floats = 128 KB
  __shared__ float red[16];
  __shared__ float s_mu, s_rs;

  const int b    = blockIdx.x;
  const int tid  = threadIdx.x;
  const int lane = tid & 31;
  const int w    = tid >> 5;
  float* lat = out + LAT_OFF + (size_t)b * (H_DIM * F_DIM);

  // Pass 1: load raw latent row, accumulate sum / sumsq
  float s = 0.f, ss = 0.f;
  const float4* lat4 = reinterpret_cast<const float4*>(lat);
  float4* s4 = reinterpret_cast<float4*>(srow);
  for (int i = tid; i < (H_DIM * F_DIM) / 4; i += 256) {
    const float4 v = lat4[i];
    s4[i] = v;
    s  += v.x + v.y + v.z + v.w;
    ss += v.x * v.x + v.y * v.y + v.z * v.z + v.w * v.w;
  }
#pragma unroll
  for (int o = 16; o > 0; o >>= 1) { s += __shfl_down(s, o); ss += __shfl_down(ss, o); }
  if (lane == 0) { red[w] = s; red[8 + w] = ss; }
  __syncthreads();
  if (tid == 0) {
    float S = 0.f, SS = 0.f;
    for (int i = 0; i < 8; ++i) { S += red[i]; SS += red[8 + i]; }
    const float inv_n = 1.0f / (float)(H_DIM * F_DIM);
    const float mu = S * inv_n;
    const float var = SS * inv_n - mu * mu;
    s_mu = mu;
    s_rs = rsqrtf(var + 1e-5f);
  }
  __syncthreads();

  // Pass 2: normalize + affine + ReLU; keep in LDS, write final latent out
  const float mu = s_mu, rs = s_rs;
  for (int i = tid; i < H_DIM * F_DIM; i += 256) {
    float v = (srow[i] - mu) * rs * ln_w[i] + ln_b[i];
    v = fmaxf(v, 0.f);
    srow[i] = v;
    lat[i] = v;
  }
  __syncthreads();

  // Pass 3a: categorical softmax heads — wave w handles head h=w.
  // lanes split: c = lane&15 (class), fh = lane>>4 (feature half)
  {
    const int h  = w;
    const int c  = lane & 15;
    const int fh = lane >> 4;
    const float* lr = srow + h * F_DIM;
    const float* wc = Wc + (size_t)h * F_DIM * CATD;
    float p = 0.f;
    for (int f = fh * (F_DIM / 2); f < (fh + 1) * (F_DIM / 2); ++f)
      p = fmaf(lr[f], wc[(size_t)f * CATD + c], p);
    p += __shfl_xor(p, 16);
    const float logit = p + bc[h * CATD + c];
    float m = logit;
#pragma unroll
    for (int o = 8; o > 0; o >>= 1) m = fmaxf(m, __shfl_xor(m, o));
    const float e = __expf(logit - m);
    float t = e;
#pragma unroll
    for (int o = 8; o > 0; o >>= 1) t += __shfl_xor(t, o);
    if (lane < 16)
      out[CAT_OFF + (size_t)b * (NCAT * CATD) + h * CATD + c] = e / t;
  }

  // Pass 3b: continuous heads — 24 heads over 8 waves, 3 per wave
#pragma unroll
  for (int t3 = 0; t3 < 3; ++t3) {
    const int hc = w + t3 * 8;                 // 0..23
    const float* lr = srow + (NCAT + hc) * F_DIM;
    const float* wv = Wv + (size_t)hc * F_DIM;
    float p = 0.f;
    for (int f = lane; f < F_DIM; f += 32) p = fmaf(lr[f], wv[f], p);
#pragma unroll
    for (int o = 16; o > 0; o >>= 1) p += __shfl_down(p, o);
    if (lane == 0)
      out[CONT_OFF + (size_t)b * NCONT + hc] = p + bv[hc];
  }
}

extern "C" void kernel_launch(void* const* d_in, const int* in_sizes, int n_in,
                              void* d_out, int out_size, void* d_ws, size_t ws_size,
                              hipStream_t stream) {
  (void)in_sizes; (void)n_in; (void)out_size; (void)d_ws; (void)ws_size;
  const float* z   = (const float*)d_in[0];
  // d_in[1] = edge (unused by the reference math)
  const float* Wp  = (const float*)d_in[2];
  const float* bp  = (const float*)d_in[3];
  const float* lnw = (const float*)d_in[4];
  const float* lnb = (const float*)d_in[5];
  const float* Wc  = (const float*)d_in[6];
  const float* bc  = (const float*)d_in[7];
  const float* Wv  = (const float*)d_in[8];
  const float* bv  = (const float*)d_in[9];
  float* out = (float*)d_out;

  // 1) batched projection GEMM -> raw latent written into d_out latent region
  dim3 grid(F_DIM / BN, B_DIM / BM, H_DIM);   // 8 x 32 x 32 = 8192 blocks
  gemm_proj_kernel<<<grid, 256, 0, stream>>>(z, Wp, bp, out + LAT_OFF);

  // 2) fused LayerNorm + ReLU + softmax/scalar heads (in-place on latent)
  ln_head_kernel<<<B_DIM, 256, (size_t)(H_DIM * F_DIM) * sizeof(float), stream>>>(
      out, lnw, lnb, Wc, bc, Wv, bv);
}
